// HierarchicalAttention_45122926412010
// MI455X (gfx1250) — matrix operand order repaired and checked by reading, not compiled, not run on hardware
//
#include <hip/hip_runtime.h>
#include <hip/hip_bf16.h>

typedef __attribute__((ext_vector_type(16))) _Float16 v16h;
typedef __attribute__((ext_vector_type(8)))  _Float16 v8h;
typedef __attribute__((ext_vector_type(8)))  float    v8f;

#define Bn 64
#define Sn 512
#define Hn 1024
#define En 28
#define EPAD 32

// ---------------------------------------------------------------------------
// Kernel 1: tok_part[b,s] = dot(tok[b,s,:], W[0:H]).  One wave per row.
// ---------------------------------------------------------------------------
__global__ void __launch_bounds__(256)
ha_tokpart_kernel(const float* __restrict__ tok,
                  const float* __restrict__ W,
                  float* __restrict__ tok_part) {
    const int wave = threadIdx.x >> 5;
    const int lane = threadIdx.x & 31;
    const int row  = blockIdx.x * 8 + wave;          // row in [0, B*S)
    const float4* t4 = (const float4*)(tok + (size_t)row * Hn);
    const float4* w4 = (const float4*)W;
    float acc = 0.f;
#pragma unroll
    for (int i = 0; i < 8; ++i) {
        float4 t = t4[i * 32 + lane];
        float4 w = w4[i * 32 + lane];
        acc += t.x * w.x + t.y * w.y + t.z * w.z + t.w * w.w;
    }
#pragma unroll
    for (int off = 16; off > 0; off >>= 1) acc += __shfl_xor(acc, off, 32);
    if (lane == 0) tok_part[row] = acc;
}

// ---------------------------------------------------------------------------
// Kernel 2: softmax over S per (b,e); write attnT[b][e][s] as f16, E padded
// to 32 rows (rows 28..31 zero).  One block per batch, LDS-staged operands.
// ---------------------------------------------------------------------------
__global__ void __launch_bounds__(256)
ha_softmax_kernel(const float* __restrict__ tok_part,
                  const float* __restrict__ emo,
                  const float* __restrict__ mask,
                  const float* __restrict__ W,
                  const float* __restrict__ bias,
                  _Float16* __restrict__ attnT) {
    __shared__ float s_emo[Sn * En];   // 56 KB
    __shared__ float s_tp[Sn];
    __shared__ float s_mk[Sn];
    const int b   = blockIdx.x;
    const int tid = threadIdx.x;
    for (int i = tid; i < Sn * En; i += 256) s_emo[i] = emo[(size_t)b * Sn * En + i];
    for (int i = tid; i < Sn; i += 256) {
        s_tp[i] = tok_part[b * Sn + i];
        s_mk[i] = mask[b * Sn + i];
    }
    __syncthreads();
    const float w_emo = W[Hn];
    const float bb    = bias[0];
    const int wave = tid >> 5, lane = tid & 31;
    for (int e = wave; e < EPAD; e += 8) {
        _Float16* orow = attnT + ((size_t)b * EPAD + e) * Sn;
        if (e < En) {
            float mx = -3.4e38f;
            for (int s = lane; s < Sn; s += 32) {
                float l = s_tp[s] + s_emo[s * En + e] * w_emo + bb;
                mx = fmaxf(mx, l);
            }
#pragma unroll
            for (int off = 16; off > 0; off >>= 1) mx = fmaxf(mx, __shfl_xor(mx, off, 32));
            float sum = 0.f;
            for (int s = lane; s < Sn; s += 32)
                sum += __expf(s_tp[s] + s_emo[s * En + e] * w_emo + bb - mx);
#pragma unroll
            for (int off = 16; off > 0; off >>= 1) sum += __shfl_xor(sum, off, 32);
            const float inv = 1.0f / sum;
            for (int s = lane; s < Sn; s += 32) {
                float a = __expf(s_tp[s] + s_emo[s * En + e] * w_emo + bb - mx) * inv * s_mk[s];
                orow[s] = (_Float16)a;
            }
        } else {
            for (int s = lane; s < Sn; s += 32) orow[s] = (_Float16)0.0f;
        }
    }
}

// ---------------------------------------------------------------------------
// Kernel 3: out[b] (28x1024) = attnT[b] (32x512,f16) x tok[b] (512x1024,f32)
// via V_WMMA_F32_16X16X32_F16.  Each wave: 32x32 output tile = 2 M-tiles x
// 2 N-tiles (4 WMMA accumulators); A fragments reused across both N-tiles,
// and one wave's B-row loads span a full 128B cacheline (32 columns).
// Grid: 64 batches x 8 column-chunks of 128; block = 4 waves.
// No divergence before the WMMAs (EXEC all ones).
// ---------------------------------------------------------------------------
__global__ void __launch_bounds__(128)
ha_gemm_kernel(const float* __restrict__ tok,
               const _Float16* __restrict__ attnT,
               float* __restrict__ out) {
    const int wave = threadIdx.x >> 5;
    const int lane = threadIdx.x & 31;
    const int b    = blockIdx.x >> 3;
    const int n0   = ((blockIdx.x & 7) << 7) + (wave << 5);   // 32 cols per wave
    const int m    = lane & 15;       // A-row within tile / B-column within tile
    const int hi   = lane >> 4;

    // A fragment base: 16-bit A 16x32 layout => per lane two contiguous
    // 8-half chunks at K offsets hi*8 and 16+hi*8.
    const _Float16* a0base = attnT + ((size_t)b * EPAD + m) * Sn + hi * 8;
    const _Float16* a1base = a0base + (size_t)16 * Sn;
    // B fragment bases: lane holds column (n0|n0+16)+m, K rows hi*8 + j (j<8), 16 + hi*8 + (j-8) (j>=8).
    const float* bbase0 = tok + (size_t)b * Sn * Hn + (size_t)(hi * 8) * Hn + (n0 + m);
    const float* bbase1 = bbase0 + 16;

    v8f acc00 = {};   // (e 0..15)  x (n0..n0+15)
    v8f acc01 = {};   // (e 0..15)  x (n0+16..n0+31)
    v8f acc10 = {};   // (e 16..31) x (n0..n0+15)
    v8f acc11 = {};   // (e 16..31) x (n0+16..n0+31)
    for (int k0 = 0; k0 < Sn; k0 += 32) {
        const float* bp0 = bbase0 + (size_t)k0 * Hn;
        const float* bp1 = bbase1 + (size_t)k0 * Hn;
        if (k0 + 32 < Sn) {
            // WGP-scope (near) prefetch of next K-tile -> global_prefetch_b8
            __builtin_prefetch((const void*)(bp0 + 32 * Hn), 0, 3);
            __builtin_prefetch((const void*)(bp1 + 32 * Hn), 0, 3);
        }

        v8h a0lo = *(const v8h*)(a0base + k0);
        v8h a0hi = *(const v8h*)(a0base + k0 + 16);
        v8h a1lo = *(const v8h*)(a1base + k0);
        v8h a1hi = *(const v8h*)(a1base + k0 + 16);
        v16h A0, A1, Bt0, Bt1;
#pragma unroll
        for (int j = 0; j < 8; ++j) {
            A0[j] = a0lo[j]; A0[j + 8] = a0hi[j];
            A1[j] = a1lo[j]; A1[j + 8] = a1hi[j];
        }
#pragma unroll
        for (int j = 0; j < 16; ++j) {
            Bt0[j] = (_Float16)bp0[(size_t)(j + (j & 8)) * Hn];
            Bt1[j] = (_Float16)bp1[(size_t)(j + (j & 8)) * Hn];
        }

        acc00 = __builtin_amdgcn_wmma_f32_16x16x32_f16(false, A0, false, Bt0,
                                                       (short)0, acc00, false, false);
        acc10 = __builtin_amdgcn_wmma_f32_16x16x32_f16(false, A1, false, Bt0,
                                                       (short)0, acc10, false, false);
        acc01 = __builtin_amdgcn_wmma_f32_16x16x32_f16(false, A0, false, Bt1,
                                                       (short)0, acc01, false, false);
        acc11 = __builtin_amdgcn_wmma_f32_16x16x32_f16(false, A1, false, Bt1,
                                                       (short)0, acc11, false, false);
    }

    // C/D layout: lanes 0-15 N=lane, VGPR v -> M=v; lanes 16-31 N=lane-16, M=8+v.
    const int nb    = lane & 15;
    const int mbase = hi * 8;
#pragma unroll
    for (int v = 0; v < 8; ++v) {
        const int e0 = mbase + v;                 // 0..15, always valid
        float* r0 = out + ((size_t)b * En + e0) * Hn + n0 + nb;
        r0[0]  = acc00[v];
        r0[16] = acc01[v];
        const int e1 = 16 + mbase + v;            // 16..31, keep < 28
        if (e1 < En) {
            float* r1 = out + ((size_t)b * En + e1) * Hn + n0 + nb;
            r1[0]  = acc10[v];
            r1[16] = acc11[v];
        }
    }
}

// ---------------------------------------------------------------------------
extern "C" void kernel_launch(void* const* d_in, const int* in_sizes, int n_in,
                              void* d_out, int out_size, void* d_ws, size_t ws_size,
                              hipStream_t stream) {
    const float* tok  = (const float*)d_in[0];   // (64,512,1024) f32
    const float* emo  = (const float*)d_in[1];   // (64,512,28)   f32
    const float* mask = (const float*)d_in[2];   // (64,512)      f32
    const float* W    = (const float*)d_in[3];   // (1,1025)      f32
    const float* bias = (const float*)d_in[4];   // (1,)          f32
    float* out = (float*)d_out;                  // (64,28,1024)  f32

    float*    tok_part = (float*)d_ws;                              // 128 KB
    _Float16* attnT    = (_Float16*)((char*)d_ws + (size_t)Bn * Sn * sizeof(float)); // 2 MiB

    ha_tokpart_kernel<<<dim3((Bn * Sn) / 8), dim3(256), 0, stream>>>(tok, W, tok_part);
    ha_softmax_kernel<<<dim3(Bn), dim3(256), 0, stream>>>(tok_part, emo, mask, W, bias, attnT);
    ha_gemm_kernel<<<dim3(Bn * 8), dim3(128), 0, stream>>>(tok, attnT, out);
}
